// HydraScaleLM_8572754723176
// MI455X (gfx1250) — matrix operand order repaired and checked by reading, not compiled, not run on hardware
//
#include <hip/hip_runtime.h>
#include <math.h>
#include <stdint.h>

// ---- model dims ----
#define DD     512
#define DEPTH  6
#define NST    16
#define DCONV  4
#define RLOW   32
#define DFF    2048
#define BB     2
#define LL     1024
#define MTOK   (BB*LL)                 // 2048
#define PROJN  (2*DD + RLOW + 2*NST)   // 1088
#define NVOCAB 50000

typedef __attribute__((ext_vector_type(16))) __bf16 bf16x16;
typedef __attribute__((ext_vector_type(8)))  float  f32x8;

union ABf16 { bf16x16 v; uint4 q[2]; };

__device__ __forceinline__ uint16_t f32_to_bf16(float f) {
  union { float f; uint32_t u; } c; c.f = f;
  uint32_t u = c.u;
  return (uint16_t)((u + 0x7FFFu + ((u >> 16) & 1u)) >> 16);
}

// ---------------- weight / activation fp32 -> bf16 convert ----------------
__global__ void cvt_bf16_kernel(const float* __restrict__ s, uint16_t* __restrict__ d, size_t n) {
  size_t i = (size_t)blockIdx.x * blockDim.x + threadIdx.x;
  size_t stride = (size_t)gridDim.x * blockDim.x;
  for (; i < n; i += stride) d[i] = f32_to_bf16(s[i]);
}

// dtu slice [M,32] out of proj [M,1088] -> bf16
__global__ void cvt_dtu_kernel(const float* __restrict__ proj, uint16_t* __restrict__ dtu) {
  int m = blockIdx.x;
  int c = threadIdx.x; // 32
  dtu[(size_t)m * RLOW + c] = f32_to_bf16(proj[(size_t)m * PROJN + 2 * DD + c]);
}

// ---------------- time embedding MLP (tiny, B=2) ----------------
__global__ __launch_bounds__(256)
void time_embed_kernel(const int* __restrict__ t,
                       const float* __restrict__ w1, const float* __restrict__ b1,
                       const float* __restrict__ w2, const float* __restrict__ b2,
                       float* __restrict__ te) {
  __shared__ float s_in[DD];
  __shared__ float s_h[DFF];
  int b = blockIdx.x;
  float tv = (float)t[b];
  int i = threadIdx.x; // 0..255 == half dim
  float f = __expf((float)i * (-logf(10000.0f) / 255.0f));
  float e = tv * f;
  s_in[i]       = __sinf(e);
  s_in[i + 256] = __cosf(e);
  __syncthreads();
  for (int j = threadIdx.x; j < DFF; j += 256) {
    const float* wr = w1 + (size_t)j * DD;
    float acc = b1[j];
    for (int k = 0; k < DD; ++k) acc = fmaf(s_in[k], wr[k], acc);
    s_h[j] = acc / (1.0f + __expf(-acc));      // silu
  }
  __syncthreads();
  for (int o = threadIdx.x; o < DD; o += 256) {
    const float* wr = w2 + (size_t)o * DFF;
    float acc = b2[o];
    for (int k = 0; k < DFF; ++k) acc = fmaf(s_h[k], wr[k], acc);
    te[b * DD + o] = acc;
  }
}

// ---------------- token embedding + time embedding ----------------
__global__ void embed_kernel(const int* __restrict__ xt, const float* __restrict__ tok,
                             const float* __restrict__ te, float* __restrict__ x) {
  int m = blockIdx.x;            // 0..2047
  int b = m / LL;
  const float* src = tok + (size_t)xt[m] * DD;
  const float* tv  = te + b * DD;
  float* dst = x + (size_t)m * DD;
  for (int d = threadIdx.x; d < DD; d += blockDim.x) dst[d] = src[d] + tv[d];
}

// ---------------- layernorm -> bf16 ----------------
__global__ __launch_bounds__(256)
void layernorm_bf16_kernel(const float* __restrict__ x, const float* __restrict__ g,
                           const float* __restrict__ bta, uint16_t* __restrict__ out) {
  __shared__ float red0[8], red1[8];
  int m = blockIdx.x;
  const float* xr = x + (size_t)m * DD;
  float v0 = xr[threadIdx.x];
  float v1 = xr[threadIdx.x + 256];
  float s  = v0 + v1;
  float s2 = v0 * v0 + v1 * v1;
  for (int off = 16; off; off >>= 1) { s += __shfl_down(s, off); s2 += __shfl_down(s2, off); }
  int lane = threadIdx.x & 31, wv = threadIdx.x >> 5;
  if (lane == 0) { red0[wv] = s; red1[wv] = s2; }
  __syncthreads();
  if (wv == 0) {
    float a  = (lane < 8) ? red0[lane] : 0.f;
    float a2 = (lane < 8) ? red1[lane] : 0.f;
    for (int off = 4; off; off >>= 1) { a += __shfl_down(a, off); a2 += __shfl_down(a2, off); }
    if (lane == 0) { red0[0] = a; red1[0] = a2; }
  }
  __syncthreads();
  float mu  = red0[0] * (1.0f / DD);
  float var = red1[0] * (1.0f / DD) - mu * mu;
  float inv = rsqrtf(var + 1e-5f);
  size_t base = (size_t)m * DD;
  out[base + threadIdx.x]       = f32_to_bf16((v0 - mu) * inv * g[threadIdx.x] + bta[threadIdx.x]);
  out[base + threadIdx.x + 256] = f32_to_bf16((v1 - mu) * inv * g[threadIdx.x + 256] + bta[threadIdx.x + 256]);
}

// ---------------- epilogue helper ----------------
__device__ __forceinline__ void gemm_epilogue_tile(
    const f32x8& acc, int m_tile, int n_tile, int half, int r16,
    const float* bias, const float* addsrc, void* out, int N, int act, int out_bf16) {
  const int col = n_tile * 16 + r16;
  const float bv = bias ? bias[col] : 0.0f;
#pragma unroll
  for (int r = 0; r < 8; ++r) {
    const int orow = m_tile * 16 + r + half * 8;
    float v = acc[r] + bv;
    if (act == 1)      v = 0.5f * v * (1.0f + erff(v * 0.70710678118f));   // gelu exact
    else if (act == 2) v = fmaxf(v, 0.0f) + log1pf(__expf(-fabsf(v)));     // softplus
    const size_t oidx = (size_t)orow * N + col;
    if (addsrc) v += addsrc[oidx];
    if (out_bf16) ((uint16_t*)out)[oidx] = f32_to_bf16(v);
    else          ((float*)out)[oidx] = v;
  }
}

// ---------------- register-path bf16 WMMA GEMM (used for K=32 and ragged N tail) ----------------
// block = 128 (4 wave32); wave w -> m_tile = blockIdx.y*4+w ; wave computes 16 x (16*NT).
template <int NT>
__global__ __launch_bounds__(128)
void gemm_bf16_nt(const uint16_t* __restrict__ A, int lda,
                  const uint16_t* __restrict__ W,
                  const float* __restrict__ bias,
                  const float* __restrict__ addsrc,
                  void* __restrict__ out,
                  int M, int N, int K, int n_tile_base, int act, int out_bf16) {
  const int lane   = threadIdx.x & 31;
  const int wave   = threadIdx.x >> 5;
  const int m_tile = blockIdx.y * 4 + wave;
  const int n_tile0 = n_tile_base + blockIdx.x * NT;
  const int r16  = lane & 15;
  const int half = lane >> 4;

  const uint16_t* aP = A + (size_t)(m_tile * 16 + r16) * lda + half * 8;
  const uint16_t* wP[NT];
#pragma unroll
  for (int j = 0; j < NT; ++j)
    wP[j] = W + (size_t)((n_tile0 + j) * 16 + r16) * K + half * 8;

  f32x8 acc[NT] = {};
#pragma unroll 2
  for (int k0 = 0; k0 < K; k0 += 32) {
    ABf16 a;
    a.q[0] = *(const uint4*)(aP + k0);
    a.q[1] = *(const uint4*)(aP + k0 + 16);
    ABf16 b[NT];
#pragma unroll
    for (int j = 0; j < NT; ++j) {
      b[j].q[0] = *(const uint4*)(wP[j] + k0);
      b[j].q[1] = *(const uint4*)(wP[j] + k0 + 16);
    }
#pragma unroll
    for (int j = 0; j < NT; ++j)
      acc[j] = __builtin_amdgcn_wmma_f32_16x16x32_bf16(
          false, a.v, false, b[j].v, (short)0, acc[j], false, false);
  }
#pragma unroll
  for (int j = 0; j < NT; ++j)
    gemm_epilogue_tile(acc[j], m_tile, n_tile0 + j, half, r16, bias, addsrc, out, N, act, out_bf16);
}

// ---------------- LDS-staged bf16 WMMA GEMM (K % 64 == 0) ----------------
// Block = 128 threads (4 waves) computing a 64M x 64N tile. The 64x64 (KCx64N) W slice is
// staged into LDS once per K-block with async load-to-LDS (ASYNCcnt), double buffered:
// stage(next) -> s_wait_asynccnt 4 (cur done, next in flight) -> barrier -> WMMA from LDS.
// This dedupes the 4x redundant per-wave W fetches of the register-path kernel.
#define KC 64
__global__ __launch_bounds__(128)
void gemm_bf16_lds(const uint16_t* __restrict__ A, int lda,
                   const uint16_t* __restrict__ W,
                   const float* __restrict__ bias,
                   const float* __restrict__ addsrc,
                   void* __restrict__ out,
                   int M, int N, int K, int act, int out_bf16) {
  __shared__ uint16_t sW[2][64 * KC];   // 2 x 8KB ; row = n (0..63), col = k (0..63)
  const int lane   = threadIdx.x & 31;
  const int wave   = threadIdx.x >> 5;
  const int m_tile = blockIdx.y * 4 + wave;
  const int nb0    = blockIdx.x * 64;    // first output column of this block
  const int r16  = lane & 15;
  const int half = lane >> 4;

  const uint16_t* aP = A + (size_t)(m_tile * 16 + r16) * lda + half * 8;

  // staging map: 128 threads x 4 x 16B chunks cover 64 rows x 128B
  const int t      = threadIdx.x;
  const int srow   = t >> 1;          // 0..63 (each row covered by 2 threads)
  const int scol   = (t & 1) * 32;    // element offset within row: 0 or 32 (+ i*8)
  const uint16_t* gstage = W + (size_t)(nb0 + srow) * K + scol;
  const uint32_t  lstage0 = (uint32_t)(uintptr_t)&sW[0][srow * KC + scol];
  const uint32_t  lstage1 = (uint32_t)(uintptr_t)&sW[1][srow * KC + scol];

  f32x8 acc[4] = {};

  // prologue: stage k-block 0 into buffer 0
  {
#pragma unroll
    for (int i = 0; i < 4; ++i) {
      uint64_t ga = (uint64_t)(uintptr_t)(gstage + i * 8);
      uint32_t la = lstage0 + i * 16;
      asm volatile("global_load_async_to_lds_b128 %0, %1, off" :: "v"(la), "v"(ga) : "memory");
    }
  }

  int buf = 0;
  for (int k0 = 0; k0 < K; k0 += KC) {
    const bool more = (k0 + KC) < K;
    if (more) {                           // stage next k-block into the other buffer
      uint32_t lb = buf ? lstage0 : lstage1;
#pragma unroll
      for (int i = 0; i < 4; ++i) {
        uint64_t ga = (uint64_t)(uintptr_t)(gstage + k0 + KC + i * 8);
        uint32_t la = lb + i * 16;
        asm volatile("global_load_async_to_lds_b128 %0, %1, off" :: "v"(la), "v"(ga) : "memory");
      }
      asm volatile("s_wait_asynccnt 0x4" ::: "memory");   // current buffer's 4 loads done
    } else {
      asm volatile("s_wait_asynccnt 0x0" ::: "memory");   // drain
    }
    __syncthreads();                                       // all waves finished staging cur

    const uint16_t* sbase = &sW[buf][0];
#pragma unroll
    for (int kk = 0; kk < KC; kk += 32) {
      ABf16 a;
      a.q[0] = *(const uint4*)(aP + k0 + kk);
      a.q[1] = *(const uint4*)(aP + k0 + kk + 16);
#pragma unroll
      for (int j = 0; j < 4; ++j) {
        const uint16_t* ls = sbase + (j * 16 + r16) * KC + kk + half * 8;
        ABf16 b;
        b.q[0] = *(const uint4*)(ls);
        b.q[1] = *(const uint4*)(ls + 16);
        acc[j] = __builtin_amdgcn_wmma_f32_16x16x32_bf16(
            false, a.v, false, b.v, (short)0, acc[j], false, false);
      }
    }
    __syncthreads();                                       // done reading cur before overwrite
    buf ^= 1;
  }

#pragma unroll
  for (int j = 0; j < 4; ++j)
    gemm_epilogue_tile(acc[j], m_tile, blockIdx.x * 4 + j, half, r16, bias, addsrc, out, N, act, out_bf16);
}

// ---------------- depthwise causal conv (width 4) + silu ----------------
__global__ void conv_silu_kernel(const float* __restrict__ proj, const float* __restrict__ wconv,
                                 float* __restrict__ u) {
  int m = blockIdx.x; int b = m / LL; int l = m % LL;
  for (int d = threadIdx.x; d < DD; d += blockDim.x) {
    float acc = 0.f;
#pragma unroll
    for (int j = 0; j < DCONV; ++j) {
      int ls = l - (DCONV - 1) + j;
      if (ls >= 0) acc += wconv[d * DCONV + j] * proj[(size_t)(b * LL + ls) * PROJN + d];
    }
    u[(size_t)m * DD + d] = acc / (1.0f + __expf(-acc));
  }
}

// ---------------- sequential SSM scan: one (b,d,n) state per lane; 16-lane reduce over n ----------------
__global__ __launch_bounds__(256)
void ssm_scan_kernel(const float* __restrict__ proj, const float* __restrict__ dt,
                     const float* __restrict__ u, float* __restrict__ y) {
  int tid = blockIdx.x * blockDim.x + threadIdx.x;  // 0..16383
  int n = tid & (NST - 1);
  int d = (tid >> 4) & (DD - 1);
  int b = tid >> 13;
  float An   = -(float)(n + 1);
  float invA = 1.0f / (An + 1e-10f);
  float h = 0.0f;
  for (int l = 0; l < LL; ++l) {
    size_t m = (size_t)b * LL + l;
    float dtv = dt[m * DD + d];
    float uv  = u [m * DD + d];
    float Bv  = proj[m * PROJN + 2 * DD + RLOW + n];
    float Cv  = proj[m * PROJN + 2 * DD + RLOW + NST + n];
    float at  = __expf(dtv * An);
    float bu  = (at - 1.0f) * invA * Bv * uv;
    h = at * h + bu;
    float c = Cv * h;
    c += __shfl_xor(c, 1, 16);
    c += __shfl_xor(c, 2, 16);
    c += __shfl_xor(c, 4, 16);
    c += __shfl_xor(c, 8, 16);
    if (n == 0) y[m * DD + d] = c;
  }
}

// ---------------- gate: yg = (y + u*Dp) * silu(z), write bf16 ----------------
__global__ void gate_kernel(const float* __restrict__ y, const float* __restrict__ u,
                            const float* __restrict__ proj, const float* __restrict__ Dp,
                            uint16_t* __restrict__ yg) {
  int m = blockIdx.x;
  for (int d = threadIdx.x; d < DD; d += blockDim.x) {
    float z = proj[(size_t)m * PROJN + DD + d];
    float v = (y[(size_t)m * DD + d] + u[(size_t)m * DD + d] * Dp[d]) * (z / (1.0f + __expf(-z)));
    yg[(size_t)m * DD + d] = f32_to_bf16(v);
  }
}

// ===================================================================================
extern "C" void kernel_launch(void* const* d_in, const int* in_sizes, int n_in,
                              void* d_out, int out_size, void* d_ws, size_t ws_size,
                              hipStream_t stream) {
  const int*   x_t     = (const int*)  d_in[0];
  const int*   t_in    = (const int*)  d_in[1];
  const float* tok_emb = (const float*)d_in[2];
  const float* tmlp_w1 = (const float*)d_in[3];
  const float* tmlp_b1 = (const float*)d_in[4];
  const float* tmlp_w2 = (const float*)d_in[5];
  const float* tmlp_b2 = (const float*)d_in[6];
  const float* ln1_g   = (const float*)d_in[7];
  const float* ln1_b   = (const float*)d_in[8];
  const float* in_w    = (const float*)d_in[9];
  const float* conv_w  = (const float*)d_in[10];
  const float* dt_w    = (const float*)d_in[11];
  const float* dt_b    = (const float*)d_in[12];
  const float* D_p     = (const float*)d_in[13];
  const float* out_w   = (const float*)d_in[14];
  const float* ln2_g   = (const float*)d_in[15];
  const float* ln2_b   = (const float*)d_in[16];
  const float* mlp_w1  = (const float*)d_in[17];
  const float* mlp_b1  = (const float*)d_in[18];
  const float* mlp_w2  = (const float*)d_in[19];
  const float* mlp_b2  = (const float*)d_in[20];
  const float* lno_g   = (const float*)d_in[21];
  const float* lno_b   = (const float*)d_in[22];
  const float* head_w  = (const float*)d_in[23];
  const float* head_b  = (const float*)d_in[24];
  float* out = (float*)d_out;

  // ---- workspace carve-up (256B aligned) ----
  char* base = (char*)d_ws;
  size_t off = 0;
  auto carve = [&](size_t bytes) -> char* {
    off = (off + 255) & ~(size_t)255;
    char* p = base + off;
    off += bytes;
    return p;
  };
  float*    te_f   = (float*)   carve((size_t)BB * DD * 4);
  float*    x_f    = (float*)   carve((size_t)MTOK * DD * 4);
  uint16_t* ln_bf  = (uint16_t*)carve((size_t)MTOK * DD * 2);
  float*    proj_f = (float*)   carve((size_t)MTOK * PROJN * 4);
  float*    u_f    = (float*)   carve((size_t)MTOK * DD * 4);
  uint16_t* dtu_bf = (uint16_t*)carve((size_t)MTOK * RLOW * 2);
  float*    dt_f   = (float*)   carve((size_t)MTOK * DD * 4);
  float*    y_f    = (float*)   carve((size_t)MTOK * DD * 4);
  uint16_t* yg_bf  = (uint16_t*)carve((size_t)MTOK * DD * 2);
  uint16_t* h_bf   = (uint16_t*)carve((size_t)MTOK * DFF * 2);
  uint16_t* wbf_in   = (uint16_t*)carve((size_t)DEPTH * PROJN * DD * 2);
  uint16_t* wbf_dt   = (uint16_t*)carve((size_t)DEPTH * DD * RLOW * 2);
  uint16_t* wbf_out  = (uint16_t*)carve((size_t)DEPTH * DD * DD * 2);
  uint16_t* wbf_m1   = (uint16_t*)carve((size_t)DEPTH * DFF * DD * 2);
  uint16_t* wbf_m2   = (uint16_t*)carve((size_t)DEPTH * DD * DFF * 2);
  uint16_t* wbf_head = (uint16_t*)carve((size_t)NVOCAB * DD * 2);
  (void)ws_size; (void)n_in; (void)in_sizes; (void)out_size;

  // ---- convert all GEMM weights to bf16 (streamed once; negligible vs head GEMM) ----
  auto cvt = [&](const float* s, uint16_t* d, size_t n) {
    cvt_bf16_kernel<<<2048, 256, 0, stream>>>(s, d, n);
  };
  cvt(in_w,   wbf_in,   (size_t)DEPTH * PROJN * DD);
  cvt(dt_w,   wbf_dt,   (size_t)DEPTH * DD * RLOW);
  cvt(out_w,  wbf_out,  (size_t)DEPTH * DD * DD);
  cvt(mlp_w1, wbf_m1,   (size_t)DEPTH * DFF * DD);
  cvt(mlp_w2, wbf_m2,   (size_t)DEPTH * DD * DFF);
  cvt(head_w, wbf_head, (size_t)NVOCAB * DD);

  // ---- embeddings ----
  time_embed_kernel<<<BB, 256, 0, stream>>>(t_in, tmlp_w1, tmlp_b1, tmlp_w2, tmlp_b2, te_f);
  embed_kernel<<<MTOK, 256, 0, stream>>>(x_t, tok_emb, te_f, x_f);

  const dim3 blk(128, 1, 1);
  auto gemm_tail = [&](const uint16_t* A, int lda, const uint16_t* W, const float* bias,
                       const float* addsrc, void* o, int M, int N, int K,
                       int tile_base, int ntail, int act, int obf) {
    dim3 grid(1, M / 64, 1);
    switch (ntail) {
      case 1: gemm_bf16_nt<1><<<grid, blk, 0, stream>>>(A, lda, W, bias, addsrc, o, M, N, K, tile_base, act, obf); break;
      case 2: gemm_bf16_nt<2><<<grid, blk, 0, stream>>>(A, lda, W, bias, addsrc, o, M, N, K, tile_base, act, obf); break;
      default: gemm_bf16_nt<3><<<grid, blk, 0, stream>>>(A, lda, W, bias, addsrc, o, M, N, K, tile_base, act, obf); break;
    }
  };
  auto gemm = [&](const uint16_t* A, int lda, const uint16_t* W, const float* bias,
                  const float* addsrc, void* o, int M, int N, int K, int act, int obf) {
    const int ntiles = N / 16;          // N is always a multiple of 16 here
    if ((K % KC) == 0) {
      const int nblk = N / 64;
      if (nblk) {
        dim3 grid(nblk, M / 64, 1);
        gemm_bf16_lds<<<grid, blk, 0, stream>>>(A, lda, W, bias, addsrc, o, M, N, K, act, obf);
      }
      const int ntail = ntiles - nblk * 4;
      if (ntail) gemm_tail(A, lda, W, bias, addsrc, o, M, N, K, nblk * 4, ntail, act, obf);
    } else {
      const int nfull = ntiles / 4;
      if (nfull) {
        dim3 grid(nfull, M / 64, 1);
        gemm_bf16_nt<4><<<grid, blk, 0, stream>>>(A, lda, W, bias, addsrc, o, M, N, K, 0, act, obf);
      }
      const int rem = ntiles & 3;
      if (rem) gemm_tail(A, lda, W, bias, addsrc, o, M, N, K, nfull * 4, rem, act, obf);
    }
  };

  for (int i = 0; i < DEPTH; ++i) {
    // --- SSM branch ---
    layernorm_bf16_kernel<<<MTOK, 256, 0, stream>>>(x_f, ln1_g + i * DD, ln1_b + i * DD, ln_bf);
    gemm(ln_bf, DD, wbf_in + (size_t)i * PROJN * DD, nullptr, nullptr, proj_f,
         MTOK, PROJN, DD, 0, 0);
    conv_silu_kernel<<<MTOK, 128, 0, stream>>>(proj_f, conv_w + (size_t)i * DD * DCONV, u_f);
    cvt_dtu_kernel<<<MTOK, 32, 0, stream>>>(proj_f, dtu_bf);
    gemm(dtu_bf, RLOW, wbf_dt + (size_t)i * DD * RLOW, dt_b + i * DD, nullptr, dt_f,
         MTOK, DD, RLOW, 2 /*softplus*/, 0);
    ssm_scan_kernel<<<64, 256, 0, stream>>>(proj_f, dt_f, u_f, y_f);
    gate_kernel<<<MTOK, 128, 0, stream>>>(y_f, u_f, proj_f, D_p + i * DD, yg_bf);
    gemm(yg_bf, DD, wbf_out + (size_t)i * DD * DD, nullptr, x_f /*residual*/, x_f,
         MTOK, DD, DD, 0, 0);
    // --- MLP branch ---
    layernorm_bf16_kernel<<<MTOK, 256, 0, stream>>>(x_f, ln2_g + i * DD, ln2_b + i * DD, ln_bf);
    gemm(ln_bf, DD, wbf_m1 + (size_t)i * DFF * DD, mlp_b1 + i * DFF, nullptr, h_bf,
         MTOK, DFF, DD, 1 /*gelu*/, 1 /*bf16 out*/);
    gemm(h_bf, DFF, wbf_m2 + (size_t)i * DD * DFF, mlp_b2 + i * DD, x_f /*residual*/, x_f,
         MTOK, DD, DFF, 0, 0);
  }

  // ---- final LN + head ----
  layernorm_bf16_kernel<<<MTOK, 256, 0, stream>>>(x_f, lno_g, lno_b, ln_bf);
  gemm(ln_bf, DD, wbf_head, head_b, nullptr, out, MTOK, NVOCAB, DD, 0, 0);
}
